// MultiSimilarityLoss_48189533061227
// MI455X (gfx1250) — matrix operand order repaired and checked by reading, not compiled, not run on hardware
//
#include <hip/hip_runtime.h>
#include <stdint.h>

#define BN 4096
#define DD 128
#define NC 80

#define TILE 128
#define KT 32
#define PITCH 40   // shorts per LDS row: 80B (16B-aligned row starts)

typedef __attribute__((ext_vector_type(16))) __bf16 v16bf;
typedef __attribute__((ext_vector_type(8)))  __bf16 v8bf;
typedef __attribute__((ext_vector_type(8)))  float  v8f;

__device__ inline unsigned short f2bf(float f) {
    union { float f; unsigned u; } v; v.f = f;
    unsigned u = v.u;
    unsigned r = (u + 0x7FFFu + ((u >> 16) & 1u)) >> 16;  // round-to-nearest-even
    return (unsigned short)r;
}

// ---- CDNA5 async global->LDS copy: 64B per lane, tracked by ASYNCcnt -------
__device__ inline void async_copy64(unsigned lds_byte_addr, const void* gsrc) {
    asm volatile(
        "global_load_async_to_lds_b128 %0, %1, off\n\t"
        "global_load_async_to_lds_b128 %0, %1, off offset:16\n\t"
        "global_load_async_to_lds_b128 %0, %1, off offset:32\n\t"
        "global_load_async_to_lds_b128 %0, %1, off offset:48"
        :: "v"(lds_byte_addr), "v"(gsrc)
        : "memory");
}
__device__ inline void wait_async0() {
    asm volatile("s_wait_asynccnt 0x0" ::: "memory");
}

// ---------------- block reductions (blockDim.x threads, power of 2) ----------
__device__ inline float blockReduceSum(float v, float* s) {
    int t = threadIdx.x;
    s[t] = v; __syncthreads();
    for (int k = blockDim.x >> 1; k > 0; k >>= 1) {
        if (t < k) s[t] += s[t + k];
        __syncthreads();
    }
    float r = s[0]; __syncthreads(); return r;
}
__device__ inline float blockReduceMax(float v, float* s) {
    int t = threadIdx.x;
    s[t] = v; __syncthreads();
    for (int k = blockDim.x >> 1; k > 0; k >>= 1) {
        if (t < k) s[t] = fmaxf(s[t], s[t + k]);
        __syncthreads();
    }
    float r = s[0]; __syncthreads(); return r;
}
__device__ inline float blockReduceMin(float v, float* s) {
    int t = threadIdx.x;
    s[t] = v; __syncthreads();
    for (int k = blockDim.x >> 1; k > 0; k >>= 1) {
        if (t < k) s[t] = fminf(s[t], s[t + k]);
        __syncthreads();
    }
    float r = s[0]; __syncthreads(); return r;
}

// ---------------- bf16 WMMA GEMM with pre-transposed B ----------------------
// Computes C(MxN) = A(MxK, row-major) * B(KxN)  where caller passes
// Bt = B^T as an [N x K] row-major bf16 array.  Both LDS tiles are then plain
// row-major copies -> staged with global_load_async_to_lds_b128 (ASYNCcnt).
// out_bf16=0: C(f32) = beta*C + AB (beta in {0,1});  out_bf16=1: Cb(bf16)=AB.
// M, N multiples of 128; K multiple of 32.
__global__ __launch_bounds__(128)
void gemm_bf16_wmma(const unsigned short* __restrict__ A,
                    const unsigned short* __restrict__ Bt,
                    float* __restrict__ C,
                    unsigned short* __restrict__ Cb,
                    int M, int N, int K, float beta, int out_bf16)
{
    __shared__ __bf16 As[2][TILE * PITCH];      // [m][k]
    __shared__ __bf16 Bs[2][TILE * PITCH];      // [n][k]

    const int tid  = threadIdx.x;       // 0..127 (4 waves of 32)
    const int lane = tid & 31;
    const int wave = tid >> 5;
    const int wr   = wave >> 1;         // wave row in 2x2
    const int wc   = wave & 1;          // wave col in 2x2
    const int bRow = blockIdx.y * TILE;
    const int bCol = blockIdx.x * TILE;
    const int ln   = lane & 15;
    const int lh   = lane >> 4;

    v8f acc[4][4];
#pragma unroll
    for (int i = 0; i < 4; ++i)
#pragma unroll
        for (int j = 0; j < 4; ++j)
            acc[i][j] = (v8f){0.f,0.f,0.f,0.f,0.f,0.f,0.f,0.f};

    const unsigned ldsA0 = (unsigned)(uintptr_t)(&As[0][tid * PITCH]);
    const unsigned ldsA1 = (unsigned)(uintptr_t)(&As[1][tid * PITCH]);
    const unsigned ldsB0 = (unsigned)(uintptr_t)(&Bs[0][tid * PITCH]);
    const unsigned ldsB1 = (unsigned)(uintptr_t)(&Bs[1][tid * PITCH]);

    auto stage = [&](int buf, int k0) {
        // thread t async-copies 32 bf16 (64B) of A row (bRow+t) and Bt row (bCol+t)
        async_copy64(buf ? ldsA1 : ldsA0,
                     (const void*)(A  + (size_t)(bRow + tid) * K + k0));
        async_copy64(buf ? ldsB1 : ldsB0,
                     (const void*)(Bt + (size_t)(bCol + tid) * K + k0));
    };
    auto compute = [&](int buf) {
        const __bf16* as = &As[buf][0];
        const __bf16* bs = &Bs[buf][0];
        v16bf afrag[4], bfrag[4];
#pragma unroll
        for (int ti = 0; ti < 4; ++ti) {
            int row = wr * 64 + ti * 16 + ln;
            v8bf lo = *(const v8bf*)(as + row * PITCH + 16 * lh);
            v8bf hi = *(const v8bf*)(as + row * PITCH + 16 * lh + 8);
            v16bf a;
#pragma unroll
            for (int e = 0; e < 8; ++e) { a[e] = lo[e]; a[8 + e] = hi[e]; }
            afrag[ti] = a;
        }
#pragma unroll
        for (int tj = 0; tj < 4; ++tj) {
            int col = wc * 64 + tj * 16 + ln;
            bfrag[tj] = *(const v16bf*)(bs + col * PITCH + 16 * lh);
        }
#pragma unroll
        for (int ti = 0; ti < 4; ++ti)
#pragma unroll
            for (int tj = 0; tj < 4; ++tj)
                acc[ti][tj] = __builtin_amdgcn_wmma_f32_16x16x32_bf16(
                    false, afrag[ti], false, bfrag[tj],
                    (short)0, acc[ti][tj], false, false);
    };

    stage(0, 0);
    wait_async0();
    __syncthreads();
    int buf = 0;
    for (int k0 = 0; k0 < K; k0 += KT) {
        int kn = k0 + KT;
        if (kn < K)
            stage(buf ^ 1, kn);              // async, overlaps with compute
        if (kn + KT < K) {                   // prefetch tile-after-next
            __builtin_prefetch(A  + (size_t)(bRow + tid) * K + (kn + KT), 0, 1);
            __builtin_prefetch(Bt + (size_t)(bCol + tid) * K + (kn + KT), 0, 1);
        }
        compute(buf);
        wait_async0();                       // staged tile fully in LDS
        __syncthreads();
        buf ^= 1;
    }

#pragma unroll
    for (int ti = 0; ti < 4; ++ti) {
#pragma unroll
        for (int tj = 0; tj < 4; ++tj) {
            int col = bCol + wc * 64 + tj * 16 + ln;
#pragma unroll
            for (int r = 0; r < 8; ++r) {
                int row = bRow + wr * 64 + ti * 16 + r + 8 * lh;
                size_t idx = (size_t)row * N + col;
                float v = acc[ti][tj][r];
                if (out_bf16) {
                    Cb[idx] = f2bf(v);
                } else if (beta != 0.f) {
                    C[idx] = C[idx] + v;
                } else {
                    C[idx] = v;
                }
            }
        }
    }
}

// ---------------- tiled transposes ------------------------------------------
__global__ __launch_bounds__(256)
void transpose_bf16(const unsigned short* __restrict__ src,
                    unsigned short* __restrict__ dst)
{
    __shared__ unsigned short tile[64][65];
    int bx = blockIdx.x * 64, by = blockIdx.y * 64;
    int tx = threadIdx.x & 63, ty = threadIdx.x >> 6;
    for (int r = ty; r < 64; r += 4)
        tile[r][tx] = src[(size_t)(by + r) * BN + bx + tx];
    __syncthreads();
    for (int r = ty; r < 64; r += 4)
        dst[(size_t)(bx + r) * BN + by + tx] = tile[tx][r];
}

__global__ __launch_bounds__(256)
void transpose_f32_bf16(const float* __restrict__ src,
                        unsigned short* __restrict__ dst)
{
    __shared__ unsigned short tile[64][65];
    int bx = blockIdx.x * 64, by = blockIdx.y * 64;
    int tx = threadIdx.x & 63, ty = threadIdx.x >> 6;
    for (int r = ty; r < 64; r += 4)
        tile[r][tx] = f2bf(src[(size_t)(by + r) * BN + bx + tx]);
    __syncthreads();
    for (int r = ty; r < 64; r += 4)
        dst[(size_t)(bx + r) * BN + by + tx] = tile[tx][r];
}

// ---------------- small elementwise / reduction kernels ---------------------
__global__ void init_kernel(float* acc) {
    if (threadIdx.x < 2) acc[threadIdx.x] = 0.f;
}

__global__ void cvt_kernel(const float* __restrict__ feats,
                           unsigned short* __restrict__ Fb)
{
    int idx = blockIdx.x * blockDim.x + threadIdx.x;
    if (idx < BN * DD) Fb[idx] = f2bf(feats[idx]);
}

__global__ void cls_kernel(const int* __restrict__ labels, int* __restrict__ cls)
{
    int i = blockIdx.x * blockDim.x + threadIdx.x;
    if (i < BN) {
        int c = 0;
        for (int k = 0; k < NC; ++k)
            if (labels[i * NC + k] != 0) c = k;
        cls[i] = c;
    }
}

// per-row: ||gram_i||, max/sum for Wm softmax (diag -1e9), logsumexp for KD
__global__ __launch_bounds__(256)
void row_stats_kernel(const float* __restrict__ gram,
                      float* __restrict__ rnorm, float* __restrict__ wmaxv,
                      float* __restrict__ wsumv, float* __restrict__ lsev)
{
    __shared__ float s1[256], s2[256], s3[256];
    int i = blockIdx.x, t = threadIdx.x;
    const float* row = gram + (size_t)i * BN;
    float ss = 0.f, wmax = -__builtin_inff(), kmax = -__builtin_inff();
    for (int j = t; j < BN; j += 256) {
        float g = row[j];
        ss += g * g;
        float wv = (j == i) ? g - 1e9f : g;       // DIFF_TAU = 1
        wmax = fmaxf(wmax, wv);
        kmax = fmaxf(kmax, g);                    // KD_TAU = 1
    }
    ss   = blockReduceSum(ss, s1);
    wmax = blockReduceMax(wmax, s2);
    kmax = blockReduceMax(kmax, s3);
    float wsum = 0.f, ksum = 0.f;
    for (int j = t; j < BN; j += 256) {
        float g = row[j];
        float wv = (j == i) ? g - 1e9f : g;
        wsum += expf(wv - wmax);
        ksum += expf(g - kmax);
    }
    wsum = blockReduceSum(wsum, s1);
    ksum = blockReduceSum(ksum, s2);
    if (t == 0) {
        rnorm[i] = sqrtf(ss);
        wmaxv[i] = wmax;
        wsumv[i] = wsum;
        lsev[i]  = kmax + logf(ksum);
    }
}

// Ab = 0.99 * softmax(gram - 1e9 I)  (bf16);  X0 = 0.01 * softmax(gram) (f32)
__global__ __launch_bounds__(256)
void build_kernel(const float* __restrict__ gram,
                  const float* __restrict__ wmaxv, const float* __restrict__ wsumv,
                  const float* __restrict__ lsev,
                  unsigned short* __restrict__ Ab, float* __restrict__ X)
{
    const size_t n = (size_t)BN * BN;
    for (size_t idx = (size_t)blockIdx.x * blockDim.x + threadIdx.x; idx < n;
         idx += (size_t)gridDim.x * blockDim.x) {
        int i = (int)(idx >> 12);
        int j = (int)(idx & (BN - 1));
        float g = gram[idx];
        float wv = (i == j) ? g - 1e9f : g;
        Ab[idx] = f2bf(0.99f * expf(wv - wmaxv[i]) / wsumv[i]);
        X[idx] = 0.01f * expf(g - lsev[i]);       // (1-w) * softmax(gram)
    }
}

// multi-similarity loss, one workgroup per row
__global__ __launch_bounds__(256)
void ms_loss_kernel(const float* __restrict__ gram, const int* __restrict__ cls,
                    const float* __restrict__ rnorm, float* __restrict__ acc)
{
    __shared__ float s1[256], s2[256];
    int i = blockIdx.x, t = threadIdx.x;
    int ci = cls[i];
    float inorm = 1.0f / fmaxf(rnorm[i], 1e-12f);
    const float* row = gram + (size_t)i * BN;

    float minpos = __builtin_inff(), maxneg = -__builtin_inff();
    float cpos = 0.f, cneg = 0.f;
    for (int j = t; j < BN; j += 256) {
        float sim = row[j] * inorm;
        bool same = (cls[j] == ci);
        if (same && sim < 1.0f - 1e-5f) { minpos = fminf(minpos, sim); cpos += 1.f; }
        if (!same)                      { maxneg = fmaxf(maxneg, sim); cneg += 1.f; }
    }
    minpos = blockReduceMin(minpos, s1);
    maxneg = blockReduceMax(maxneg, s2);
    cpos   = blockReduceSum(cpos, s1);
    cneg   = blockReduceSum(cneg, s2);

    float psum = 0.f, nsum = 0.f, psel = 0.f, nsel = 0.f;
    for (int j = t; j < BN; j += 256) {
        float sim = row[j] * inorm;
        bool same = (cls[j] == ci);
        bool pos  = same && (sim < 1.0f - 1e-5f);
        if (pos && (sim - 0.1f < maxneg))   { psum += expf(-2.0f  * (sim - 0.5f)); psel += 1.f; }
        if (!same && (sim + 0.1f > minpos)) { nsum += expf(40.0f * (sim - 0.5f)); nsel += 1.f; }
    }
    psum = blockReduceSum(psum, s1);
    nsum = blockReduceSum(nsum, s2);
    psel = blockReduceSum(psel, s1);
    nsel = blockReduceSum(nsel, s2);
    if (t == 0 && cpos > 0.f && cneg > 0.f && psel > 0.f && nsel > 0.f) {
        float per = log1pf(psum) * 0.5f + log1pf(nsum) * 0.025f;
        atomicAdd(acc, per);
    }
}

// kd = sum target * (log target - log_softmax(gram))
__global__ __launch_bounds__(256)
void kd_kernel(const float* __restrict__ gram, const float* __restrict__ X,
               const float* __restrict__ lsev, float* __restrict__ acc)
{
    __shared__ float s1[256];
    const size_t n = (size_t)BN * BN;
    float local = 0.f;
    for (size_t idx = (size_t)blockIdx.x * blockDim.x + threadIdx.x; idx < n;
         idx += (size_t)gridDim.x * blockDim.x) {
        int i = (int)(idx >> 12);
        float tv = X[idx];
        float logp = gram[idx] - lsev[i];
        local += tv * (logf(fmaxf(tv, 1e-30f)) - logp);
    }
    local = blockReduceSum(local, s1);
    if (threadIdx.x == 0) atomicAdd(acc, local);
}

__global__ void finalize_kernel(const float* __restrict__ acc, float* __restrict__ out)
{
    // ms/B + (100/N_EPOCHS)*KD_LAMBDA*KD_TAU^2 * kd/B = ms/4096 + 75*kd/4096
    out[0] = acc[0] * (1.0f / 4096.0f) + 75.0f * (acc[1] * (1.0f / 4096.0f));
}

// ---------------- host-side orchestration -----------------------------------
extern "C" void kernel_launch(void* const* d_in, const int* in_sizes, int n_in,
                              void* d_out, int out_size, void* d_ws, size_t ws_size,
                              hipStream_t stream)
{
    (void)in_sizes; (void)n_in; (void)out_size; (void)ws_size;
    const float* feats  = (const float*)d_in[0];
    const int*   labels = (const int*)d_in[1];
    float* out = (float*)d_out;

    char* w = (char*)d_ws;
    auto take = [&](size_t bytes) -> char* {
        char* p = w; w += (bytes + 255) & ~(size_t)255; return p;
    };
    const size_t BB = (size_t)BN * BN;
    float*          gram = (float*)take(BB * 4);
    float*          X    = (float*)take(BB * 4);
    unsigned short* XbT  = (unsigned short*)take(BB * 2);
    unsigned short* P0   = (unsigned short*)take(BB * 2);   // Ab
    unsigned short* P1   = (unsigned short*)take(BB * 2);   // Ab^T
    unsigned short* P2   = (unsigned short*)take(BB * 2);   // spare / A^2
    unsigned short* Fb   = (unsigned short*)take((size_t)BN * DD * 2);
    int*            cls  = (int*)take((size_t)BN * 4);
    float*          rnrm = (float*)take((size_t)BN * 4);
    float*          wmx  = (float*)take((size_t)BN * 4);
    float*          wsm  = (float*)take((size_t)BN * 4);
    float*          lse  = (float*)take((size_t)BN * 4);
    float*          acc  = (float*)take(2 * 4);

    init_kernel<<<1, 64, 0, stream>>>(acc);
    cvt_kernel<<<(BN * DD + 255) / 256, 256, 0, stream>>>(feats, Fb);
    cls_kernel<<<(BN + 255) / 256, 256, 0, stream>>>(labels, cls);

    dim3 gg(BN / TILE, BN / TILE);
    dim3 gt(BN / 64, BN / 64);

    // gram = F F^T : A-role = Fb, Bt-role = (F^T)^T = Fb itself
    gemm_bf16_wmma<<<gg, 128, 0, stream>>>(Fb, Fb, gram, nullptr, BN, BN, DD, 0.f, 0);

    row_stats_kernel<<<BN, 256, 0, stream>>>(gram, rnrm, wmx, wsm, lse);
    build_kernel<<<8192, 256, 0, stream>>>(gram, wmx, wsm, lse, P0, X);
    transpose_bf16<<<gt, 256, 0, stream>>>(P0, P1);        // Ab^T
    transpose_f32_bf16<<<gt, 256, 0, stream>>>(X, XbT);    // X0^T (bf16)

    // (I - wW)^-1 via product form: X += A X ; A = A^2  (sum A^i, i < 1024)
    for (int k = 0; k < 10; ++k) {
        gemm_bf16_wmma<<<gg, 128, 0, stream>>>(P0, XbT, X, nullptr, BN, BN, BN, 1.f, 0);
        if (k < 9)
            gemm_bf16_wmma<<<gg, 128, 0, stream>>>(P0, P1, nullptr, P2, BN, BN, BN, 0.f, 1);
        transpose_f32_bf16<<<gt, 256, 0, stream>>>(X, XbT);
        if (k < 9) {
            transpose_bf16<<<gt, 256, 0, stream>>>(P2, P1);   // P1 := (A^2)^T
            unsigned short* t = P0; P0 = P2; P2 = t;          // Ab := A^2
        }
    }

    ms_loss_kernel<<<BN, 256, 0, stream>>>(gram, cls, rnrm, acc);   // -> acc[0]
    kd_kernel<<<4096, 256, 0, stream>>>(gram, X, lse, acc + 1);     // -> acc[1]
    finalize_kernel<<<1, 1, 0, stream>>>(acc, out);
}
